// StudentEncoder_61924838474345
// MI455X (gfx1250) — matrix operand order repaired and checked by reading, not compiled, not run on hardware
//
#include <hip/hip_runtime.h>

typedef __attribute__((ext_vector_type(16))) _Float16 v16h;
typedef __attribute__((ext_vector_type(8)))  _Float16 v8h;
typedef __attribute__((ext_vector_type(8)))  float    v8f;

#define BDIM 2
#define NP   1024
#define MP   1024
#define CDIM 128
#define INNERC 512
#define GG   8
#define HH   8
#define DHH  64
#define NDS  256
#define KCC  128

// ---------------------------------------------------------------------------
// Wave-per-16x64-strip WMMA GEMM:  D = A(MxK) * BT(NxK)^T  (+add +colbias)
// A row-major (K contiguous), BT row-major (K contiguous).
// One wave (32 threads) per 16x64 strip: A fragment loaded once per K-step,
// reused across 4 independent accumulators (hides WMMA RAW latency).
// EXEC all-ones (no divergence) as WMMA requires.
// ---------------------------------------------------------------------------
template <int KVAL, bool HASADD, bool HASCB>
__global__ void wmma_gemm_kernel(const _Float16* __restrict__ A,
                                 const _Float16* __restrict__ BT,
                                 float* __restrict__ D,
                                 const float* __restrict__ addsrc,
                                 const float* __restrict__ colbias,
                                 int lda, int ldb, int ldd, int batch2,
                                 long long sA1, long long sA2,
                                 long long sB1, long long sB2,
                                 long long sD1, long long sD2)
{
    int lane = threadIdx.x;
    int i0 = blockIdx.x * 16;
    int j0 = blockIdx.y * 64;
    int z  = blockIdx.z;
    int b1 = z / batch2, b2 = z % batch2;

    const _Float16* Ab = A  + b1 * sA1 + b2 * sA2;
    const _Float16* Bb = BT + b1 * sB1 + b2 * sB2;
    float*          Db = D  + b1 * sD1 + b2 * sD2;

    int m    = lane & 15;   // A row-in-tile / B,D col-in-tile
    int half = lane >> 4;

    const _Float16* arow = Ab + (size_t)(i0 + m) * lda;

    v8f c0 = {}, c1 = {}, c2 = {}, c3 = {};
#pragma unroll
    for (int kk = 0; kk < KVAL; kk += 32) {
        v16h a;
        // A 16x32 fragment: lane half selects K groups {half*8..} and {16+half*8..}
        ((v8h*)&a)[0] = *(const v8h*)(arow + kk + half * 8);
        ((v8h*)&a)[1] = *(const v8h*)(arow + kk + 16 + half * 8);
#pragma unroll
        for (int tt = 0; tt < 4; tt++) {
            v16h b;
            const _Float16* brow = Bb + (size_t)(j0 + tt * 16 + m) * ldb;
            ((v8h*)&b)[0] = *(const v8h*)(brow + kk + half * 8);
            ((v8h*)&b)[1] = *(const v8h*)(brow + kk + 16 + half * 8);
            v8f& acc = (tt == 0) ? c0 : (tt == 1) ? c1 : (tt == 2) ? c2 : c3;
            acc = __builtin_amdgcn_wmma_f32_16x16x32_f16(false, a, false, b,
                                                         (short)0, acc, false, false);
        }
    }

    const float* adds = HASADD ? (addsrc + b1 * sD1 + b2 * sD2) : nullptr;
#pragma unroll
    for (int tt = 0; tt < 4; tt++) {
        const v8f& acc = (tt == 0) ? c0 : (tt == 1) ? c1 : (tt == 2) ? c2 : c3;
        int col = j0 + tt * 16 + m;
        float cb = HASCB ? colbias[col] : 0.f;
#pragma unroll
        for (int r = 0; r < 8; r++) {
            int row = i0 + r + 8 * half;   // D layout: VGPR r -> M=r (lanes 0-15), r+8 (16-31)
            float val = acc[r] + cb;
            if (HASADD) val += adds[(size_t)row * ldd + col];
            Db[(size_t)row * ldd + col] = val;
        }
    }
}

// ---------------------------------------------------------------------------
// LayerNorm over last dim (C=128). One block (128 threads) per row.
// ---------------------------------------------------------------------------
__global__ void ln_kernel(const float* __restrict__ x, const float* __restrict__ g,
                          const float* __restrict__ bt, float* __restrict__ y)
{
    __shared__ float red[CDIM];
    int row = blockIdx.x, t = threadIdx.x;
    float v = x[(size_t)row * CDIM + t];
    red[t] = v; __syncthreads();
    for (int s = 64; s > 0; s >>= 1) { if (t < s) red[t] += red[t + s]; __syncthreads(); }
    float mean = red[0] * (1.0f / CDIM);
    __syncthreads();
    float d = v - mean;
    red[t] = d * d; __syncthreads();
    for (int s = 64; s > 0; s >>= 1) { if (t < s) red[t] += red[t + s]; __syncthreads(); }
    float var = red[0] * (1.0f / CDIM);
    y[(size_t)row * CDIM + t] = d * rsqrtf(var + 1e-5f) * g[t] + bt[t];
}

// ---------------------------------------------------------------------------
// Grouped 1x1 q projection: q[b,n,o] = sum_c x1n[b,n,(o/64)*16+c]*qw[o,c]
// Also emits scaled f16 copy (q * DH^-0.5) for the QK^T WMMA.
// ---------------------------------------------------------------------------
__global__ void qproj_kernel(const float* __restrict__ x1n, const float* __restrict__ qw,
                             float* __restrict__ q, _Float16* __restrict__ qh, float scale)
{
    __shared__ float xr[CDIM];
    int row = blockIdx.x, o = threadIdx.x;
    if (o < CDIM) xr[o] = x1n[(size_t)row * CDIM + o];
    __syncthreads();
    const float* xs = xr + (o >> 6) * 16;
    const float* w  = qw + (size_t)o * 16;
    float acc = 0.f;
#pragma unroll
    for (int cc = 0; cc < 16; cc++) acc += xs[cc] * w[cc];
    q[(size_t)row * INNERC + o]  = acc;
    qh[(size_t)row * INNERC + o] = (_Float16)(acc * scale);
}

// ---------------------------------------------------------------------------
// Offset branch: depthwise conv(k=6, s=4, pad=1) + GELU + 64->1 linear + tanh.
// ---------------------------------------------------------------------------
__global__ void offset_kernel(const float* __restrict__ q, const float* __restrict__ w1,
                              const float* __restrict__ b1v, const float* __restrict__ w2,
                              float* __restrict__ vs)
{
    int bg = blockIdx.x;       // 0..15
    int nd = threadIdx.x;      // 0..255
    int b = bg >> 3, g = bg & 7;
    int nbase = nd * 4 - 1;
    float s = 0.f;
    for (int ch = 0; ch < 64; ch++) {
        float acc = 0.f;
#pragma unroll
        for (int kk = 0; kk < 6; kk++) {
            int n = nbase + kk;
            float xv = (n >= 0 && n < NP)
                     ? q[((size_t)b * NP + n) * INNERC + g * 64 + ch] : 0.f;
            acc += xv * w1[ch * 6 + kk];
        }
        acc += b1v[ch];
        float ge = 0.5f * acc * (1.f + erff(acc * 0.70710678118654752f)); // exact GELU
        s += ge * w2[ch];
    }
    float off = tanhf(s) * 4.0f;
    float vgrid = (float)nd + off;
    vs[bg * NDS + nd] = 2.f * vgrid / (float)(NDS - 1) - 1.f;
}

// ---------------------------------------------------------------------------
// 1-D bilinear grid sample (zeros padding, align_corners=False). kv (B, Nd, C).
// ---------------------------------------------------------------------------
__global__ void gsample_kernel(const float* __restrict__ x2n, const float* __restrict__ vs,
                               float* __restrict__ kv)
{
    int idx = blockIdx.x * blockDim.x + threadIdx.x;   // (b*Nd + nd)*C + ch
    if (idx >= BDIM * NDS * CDIM) return;
    int ch = idx & (CDIM - 1);
    int nd = (idx >> 7) % NDS;
    int b  = idx / (NDS * CDIM);
    int g  = ch >> 4;
    float grid = vs[(b * GG + g) * NDS + nd];
    float ix = ((grid + 1.f) * (float)MP - 1.f) * 0.5f;
    float x0 = floorf(ix);
    float w1 = ix - x0;
    int i0 = (int)x0;
    float r = 0.f;
    if (x0 >= 0.f && x0 <= (float)(MP - 1))
        r += x2n[((size_t)b * MP + i0) * CDIM + ch] * (1.f - w1);
    if (x0 + 1.f >= 0.f && x0 + 1.f <= (float)(MP - 1)) {
        int i1 = min(max(i0 + 1, 0), MP - 1);
        r += x2n[((size_t)b * MP + i1) * CDIM + ch] * w1;
    }
    kv[idx] = r;
}

// ---------------------------------------------------------------------------
// k/v grouped 1x1 projections. k stored (B, Nd, INNER) f16 (K-contig for QK^T),
// v stored transposed (B, INNER, Nd) f16 (K=j contiguous for attn*V).
// ---------------------------------------------------------------------------
__global__ void kvproj_kernel(const float* __restrict__ kv, const float* __restrict__ kw,
                              const float* __restrict__ vw, _Float16* __restrict__ kh,
                              _Float16* __restrict__ vT)
{
    __shared__ float xr[CDIM];
    int row = blockIdx.x;                  // b*Nd + nd
    int b = row / NDS, nd = row % NDS;
    int o = threadIdx.x;
    if (o < CDIM) xr[o] = kv[(size_t)row * CDIM + o];
    __syncthreads();
    const float* xs = xr + (o >> 6) * 16;
    float ka = 0.f, va = 0.f;
#pragma unroll
    for (int cc = 0; cc < 16; cc++) {
        ka += xs[cc] * kw[o * 16 + cc];
        va += xs[cc] * vw[o * 16 + cc];
    }
    kh[(size_t)row * INNERC + o] = (_Float16)ka;
    vT[((size_t)b * INNERC + o) * NDS + nd] = (_Float16)va;
}

// ---------------------------------------------------------------------------
// CPB MLP (1->32->32->1) per (bg,i,j) with the 32x32 middle layer on WMMA.
// Block = 256 threads = 8 waves; one block per (bg,i); wave w owns j=w*32..+31.
// Per wave: h1 rows -> LDS f16, 2x2 tiles of v_wmma against w2 (LDS), fused
// relu(+b2)*w3 scatter to LDS, then per-lane row sum.
// ---------------------------------------------------------------------------
__global__ void cpb_wmma_kernel(const float* __restrict__ vs, const float* __restrict__ w1,
                                const float* __restrict__ b1v, const float* __restrict__ w2,
                                const float* __restrict__ b2v, const float* __restrict__ w3,
                                const float* __restrict__ b3v, float* __restrict__ bias)
{
    __shared__ _Float16 w2h[32 * 32];          // BT layout: row o, K=c contiguous
    __shared__ float    sw1[32], sb1[32], sb2[32], sw3[32];
    __shared__ _Float16 h1buf[8][32][32];      // per-wave A: 32 rows x 32 cols
    __shared__ float    ppbuf[8][32][32];      // per-wave relu(h2)*w3 partials

    int t = threadIdx.x;
    int wv = t >> 5, l = t & 31;
    for (int k = t; k < 1024; k += 256) w2h[k] = (_Float16)w2[k];
    if (t < 32) { sw1[t] = w1[t]; sb1[t] = b1v[t]; sb2[t] = b2v[t]; sw3[t] = w3[t]; }
    __syncthreads();

    int blk = blockIdx.x;          // bg*N + i
    int bg = blk / NP, i = blk % NP;
    int j = wv * 32 + l;

    float seq = 2.f * (float)i / (float)(NP - 1) - 1.f;
    float rel = seq - vs[bg * NDS + j];
    float sg = (rel > 0.f) ? 1.f : ((rel < 0.f) ? -1.f : 0.f);
    float xb = sg * logf(fabsf(rel) + 1.f);
#pragma unroll
    for (int c = 0; c < 32; c++)
        h1buf[wv][l][c] = (_Float16)fmaxf(xb * sw1[c] + sb1[c], 0.f);
    __syncthreads();

    int m = l & 15, half = l >> 4;
#pragma unroll
    for (int mt = 0; mt < 2; mt++) {
        v16h a;
        const _Float16* ar = &h1buf[wv][mt * 16 + m][0];
        ((v8h*)&a)[0] = *(const v8h*)(ar + half * 8);
        ((v8h*)&a)[1] = *(const v8h*)(ar + 16 + half * 8);
#pragma unroll
        for (int nt = 0; nt < 2; nt++) {
            v16h b;
            const _Float16* br = &w2h[(nt * 16 + m) * 32];
            ((v8h*)&b)[0] = *(const v8h*)(br + half * 8);
            ((v8h*)&b)[1] = *(const v8h*)(br + 16 + half * 8);
            v8f cz = {};
            cz = __builtin_amdgcn_wmma_f32_16x16x32_f16(false, a, false, b,
                                                        (short)0, cz, false, false);
            int col = nt * 16 + m;
            float wb2 = sb2[col], ww3 = sw3[col];
#pragma unroll
            for (int r = 0; r < 8; r++) {
                int row = mt * 16 + r + 8 * half;
                ppbuf[wv][row][col] = fmaxf(cz[r] + wb2, 0.f) * ww3;
            }
        }
    }
    __syncthreads();

    float out = b3v[0];
#pragma unroll
    for (int c = 0; c < 32; c++) out += ppbuf[wv][l][c];
    bias[(size_t)blk * NDS + j] = out;
}

// ---------------------------------------------------------------------------
// Softmax over Nd; writes f32 attn (into d_out) and f16 copy for attn*V WMMA.
// ---------------------------------------------------------------------------
__global__ void softmax_kernel(const float* __restrict__ sim, float* __restrict__ attn,
                               _Float16* __restrict__ attnh)
{
    __shared__ float red[NDS];
    int row = blockIdx.x, t = threadIdx.x;
    float v = sim[(size_t)row * NDS + t];
    red[t] = v; __syncthreads();
    for (int s = 128; s > 0; s >>= 1) { if (t < s) red[t] = fmaxf(red[t], red[t + s]); __syncthreads(); }
    float mx = red[0]; __syncthreads();
    float e = expf(v - mx);
    red[t] = e; __syncthreads();
    for (int s = 128; s > 0; s >>= 1) { if (t < s) red[t] += red[t + s]; __syncthreads(); }
    float r = e / red[0];
    attn [(size_t)row * NDS + t] = r;
    attnh[(size_t)row * NDS + t] = (_Float16)r;
}

__global__ void f32_to_f16_kernel(const float* __restrict__ src, _Float16* __restrict__ dst, int n)
{
    int i = blockIdx.x * blockDim.x + threadIdx.x;
    if (i < n) dst[i] = (_Float16)src[i];
}

// ---------------------------------------------------------------------------
// Cluster merge (DPC-KNN) pieces
// ---------------------------------------------------------------------------
__global__ void init_kernel(float* __restrict__ allw, float* __restrict__ xmerged,
                            int* __restrict__ dmaxi)
{
    int i = blockIdx.x * blockDim.x + threadIdx.x;
    if (i < BDIM * KCC) allw[i] = 1e-6f;
    if (i < BDIM * KCC * CDIM) xmerged[i] = 0.f;
    if (i < BDIM) dmaxi[i] = 0;
}

__global__ void rowstats_kernel(const float* __restrict__ x, const float* __restrict__ sw,
                                const float* __restrict__ sb, float* __restrict__ xx,
                                float* __restrict__ tw)
{
    int i = blockIdx.x * blockDim.x + threadIdx.x;
    if (i >= BDIM * NP) return;
    const float* r = x + (size_t)i * CDIM;
    float s2 = 0.f, sd = 0.f;
    for (int c = 0; c < CDIM; c++) { float v = r[c]; s2 += v * v; sd += v * sw[c]; }
    xx[i] = s2;
    tw[i] = expf(sd + sb[0]);
}

// dist row + density (5 smallest) + batch max via atomic
__global__ void dist_kernel(const float* __restrict__ gram, const float* __restrict__ xx,
                            float* __restrict__ dist, float* __restrict__ density,
                            float* __restrict__ dmax)
{
    __shared__ float red[256];
    __shared__ int  redi[256];
    int row = blockIdx.x;            // b*N + i
    int b = row / NP;
    int t = threadIdx.x;
    float inv = rsqrtf((float)CDIM);
    float xi = xx[row];
    float dv[4];
    float rowmax = 0.f;
#pragma unroll
    for (int u = 0; u < 4; u++) {
        int j = t * 4 + u;
        float d2 = fmaxf(xi + xx[b * NP + j] - 2.f * gram[(size_t)row * NP + j], 0.f);
        float d = sqrtf(d2 + 1e-12f) * inv;
        dv[u] = d;
        dist[(size_t)row * NP + j] = d;
        rowmax = fmaxf(rowmax, d);
    }
    red[t] = rowmax; __syncthreads();
    for (int s = 128; s > 0; s >>= 1) { if (t < s) red[t] = fmaxf(red[t], red[t + s]); __syncthreads(); }
    if (t == 0) atomicMax((int*)&dmax[b], __float_as_int(red[0]));   // dists >= 0
    // 5 smallest (incl. self) via masked min-reductions, tie -> lowest index
    bool used[4] = {false, false, false, false};
    float acc = 0.f;
    for (int it = 0; it < 5; it++) {
        float lm = 1e30f; int li = NP * 8;
#pragma unroll
        for (int u = 0; u < 4; u++)
            if (!used[u] && dv[u] < lm) { lm = dv[u]; li = t * 4 + u; }
        __syncthreads();
        red[t] = lm; redi[t] = li; __syncthreads();
        for (int s = 128; s > 0; s >>= 1) {
            if (t < s) {
                if (red[t + s] < red[t] || (red[t + s] == red[t] && redi[t + s] < redi[t])) {
                    red[t] = red[t + s]; redi[t] = redi[t + s];
                }
            }
            __syncthreads();
        }
        float gmin = red[0]; int gidx = redi[0];
        acc += gmin * gmin;
        if (gidx >= t * 4 && gidx < t * 4 + 4) used[gidx - t * 4] = true;
        __syncthreads();
    }
    if (t == 0) density[row] = expf(-acc * 0.2f);
}

__global__ void distind_kernel(const float* __restrict__ dist, const float* __restrict__ density,
                               const float* __restrict__ dmax, float* __restrict__ score)
{
    __shared__ float red[256];
    int row = blockIdx.x; int b = row / NP; int t = threadIdx.x;
    float di = density[row];
    float dm = dmax[b];
    float lm = 1e30f;
#pragma unroll
    for (int u = 0; u < 4; u++) {
        int j = t * 4 + u;
        float v = (density[b * NP + j] > di) ? dist[(size_t)row * NP + j] : dm;
        lm = fminf(lm, v);
    }
    red[t] = lm; __syncthreads();
    for (int s = 128; s > 0; s >>= 1) { if (t < s) red[t] = fminf(red[t], red[t + s]); __syncthreads(); }
    if (t == 0) score[row] = red[0] * di;
}

// iterative top-KC per batch (stable: ties -> lowest index)
__global__ void topk_kernel(const float* __restrict__ score, int* __restrict__ idx_down)
{
    __shared__ float red[256];
    __shared__ int  redi[256];
    __shared__ float sc[NP];
    int b = blockIdx.x, t = threadIdx.x;
    for (int j = t; j < NP; j += 256) sc[j] = score[b * NP + j];
    __syncthreads();
    for (int c = 0; c < KCC; c++) {
        float lm = -1e30f; int li = NP;
#pragma unroll
        for (int u = 0; u < 4; u++) {
            int j = t * 4 + u;
            float v = sc[j];
            if (v > lm || (v == lm && j < li)) { lm = v; li = j; }
        }
        red[t] = lm; redi[t] = li; __syncthreads();
        for (int s = 128; s > 0; s >>= 1) {
            if (t < s) {
                if (red[t + s] > red[t] || (red[t + s] == red[t] && redi[t + s] < redi[t])) {
                    red[t] = red[t + s]; redi[t] = redi[t + s];
                }
            }
            __syncthreads();
        }
        if (t == 0) { idx_down[b * KCC + c] = redi[0]; sc[redi[0]] = -1e30f; }
        __syncthreads();
    }
}

__global__ void assign_kernel(const float* __restrict__ dist, const int* __restrict__ idx_down,
                              int* __restrict__ cluster)
{
    int idx = blockIdx.x * blockDim.x + threadIdx.x;    // b*N + i
    if (idx >= BDIM * NP) return;
    int b = idx / NP;
    float best = 1e30f; int bc = 0;
    for (int c = 0; c < KCC; c++) {
        float d = dist[(size_t)idx * NP + idx_down[b * KCC + c]];   // dist symmetric
        if (d < best) { best = d; bc = c; }
    }
    cluster[idx] = bc;
}

__global__ void center_override_kernel(const int* __restrict__ idx_down, int* __restrict__ cluster)
{
    int idx = blockIdx.x * blockDim.x + threadIdx.x;    // b*KC + c
    if (idx >= BDIM * KCC) return;
    int b = idx / KCC, c = idx % KCC;
    cluster[b * NP + idx_down[idx]] = c;
}

__global__ void accw_kernel(const int* __restrict__ cluster, const float* __restrict__ tw,
                            float* __restrict__ allw)
{
    int idx = blockIdx.x * blockDim.x + threadIdx.x;
    if (idx >= BDIM * NP) return;
    int b = idx / NP;
    atomicAdd(&allw[b * KCC + cluster[idx]], tw[idx]);
}

__global__ void merge_kernel(const float* __restrict__ x, const int* __restrict__ cluster,
                             const float* __restrict__ tw, const float* __restrict__ allw,
                             float* __restrict__ xmerged)
{
    int row = blockIdx.x;        // b*N + i
    int b = row / NP;
    int ch = threadIdx.x;        // 128
    int cl = cluster[row];
    float w = tw[row] / allw[b * KCC + cl];
    atomicAdd(&xmerged[((size_t)b * KCC + cl) * CDIM + ch], x[(size_t)row * CDIM + ch] * w);
}

// ---------------------------------------------------------------------------
extern "C" void kernel_launch(void* const* d_in, const int* in_sizes, int n_in,
                              void* d_out, int out_size, void* d_ws, size_t ws_size,
                              hipStream_t stream)
{
    const float* x1      = (const float*)d_in[0];
    const float* x2      = (const float*)d_in[1];
    const float* ln_g    = (const float*)d_in[2];
    const float* ln_b    = (const float*)d_in[3];
    const float* q_w     = (const float*)d_in[4];
    const float* k_w     = (const float*)d_in[5];
    const float* v_w     = (const float*)d_in[6];
    const float* out_w   = (const float*)d_in[7];
    const float* out_b   = (const float*)d_in[8];
    const float* off_w1  = (const float*)d_in[9];
    const float* off_b1  = (const float*)d_in[10];
    const float* off_w2  = (const float*)d_in[11];
    const float* cpb_w1  = (const float*)d_in[12];
    const float* cpb_b1  = (const float*)d_in[13];
    const float* cpb_w2  = (const float*)d_in[14];
    const float* cpb_b2  = (const float*)d_in[15];
    const float* cpb_w3  = (const float*)d_in[16];
    const float* cpb_b3  = (const float*)d_in[17];
    const float* score_w = (const float*)d_in[18];
    const float* score_b = (const float*)d_in[19];
    (void)in_sizes; (void)n_in; (void)out_size; (void)ws_size;

    float* xmerged = (float*)d_out;                     // (B, KC, C)
    float* attn    = (float*)d_out + BDIM * KCC * CDIM; // (B, H, N, Nd) f32

    char* wsp = (char*)d_ws;
    size_t off = 0;
    auto alloc = [&](size_t bytes) {
        void* p = wsp + off;
        off = (off + bytes + 255) & ~(size_t)255;
        return p;
    };
    float*    x1n    = (float*)   alloc((size_t)BDIM * NP * CDIM * 4);
    float*    x2n    = (float*)   alloc((size_t)BDIM * MP * CDIM * 4);
    float*    q      = (float*)   alloc((size_t)BDIM * NP * INNERC * 4);
    _Float16* qh     = (_Float16*)alloc((size_t)BDIM * NP * INNERC * 2);
    float*    vs     = (float*)   alloc((size_t)BDIM * GG * NDS * 4);
    float*    kv     = (float*)   alloc((size_t)BDIM * NDS * CDIM * 4);
    _Float16* kh     = (_Float16*)alloc((size_t)BDIM * NDS * INNERC * 2);
    _Float16* vT     = (_Float16*)alloc((size_t)BDIM * INNERC * NDS * 2);
    float*    biasb  = (float*)   alloc((size_t)BDIM * GG * NP * NDS * 4);
    float*    sim    = (float*)   alloc((size_t)BDIM * HH * NP * NDS * 4);
    _Float16* attnh  = (_Float16*)alloc((size_t)BDIM * HH * NP * NDS * 2);
    float*    out2   = (float*)   alloc((size_t)BDIM * NP * INNERC * 4);
    _Float16* out2h  = (_Float16*)alloc((size_t)BDIM * NP * INNERC * 2);
    _Float16* outwh  = (_Float16*)alloc((size_t)CDIM * INNERC * 2);
    float*    xbuf   = (float*)   alloc((size_t)BDIM * NP * CDIM * 4);
    _Float16* xh     = (_Float16*)alloc((size_t)BDIM * NP * CDIM * 2);
    float*    gram   = (float*)   alloc((size_t)BDIM * NP * NP * 4);
    float*    dist   = (float*)   alloc((size_t)BDIM * NP * NP * 4);
    float*    xx     = (float*)   alloc((size_t)BDIM * NP * 4);
    float*    tw     = (float*)   alloc((size_t)BDIM * NP * 4);
    float*    density= (float*)   alloc((size_t)BDIM * NP * 4);
    float*    dmax   = (float*)   alloc((size_t)BDIM * 4);
    float*    score  = (float*)   alloc((size_t)BDIM * NP * 4);
    int*      idxdn  = (int*)     alloc((size_t)BDIM * KCC * 4);
    int*      cluster= (int*)     alloc((size_t)BDIM * NP * 4);
    float*    allw   = (float*)   alloc((size_t)BDIM * KCC * 4);

    // init (allw=1e-6, xmerged=0, dmax=0)
    init_kernel<<<(BDIM * KCC * CDIM + 255) / 256, 256, 0, stream>>>(allw, xmerged, (int*)dmax);

    // layernorms
    ln_kernel<<<BDIM * NP, CDIM, 0, stream>>>(x1, ln_g, ln_b, x1n);
    ln_kernel<<<BDIM * MP, CDIM, 0, stream>>>(x2, ln_g, ln_b, x2n);

    // q projection (+ scaled f16)
    qproj_kernel<<<BDIM * NP, INNERC, 0, stream>>>(x1n, q_w, q, qh, 0.125f);

    // offsets -> normalized sampling grid vs
    offset_kernel<<<BDIM * GG, NDS, 0, stream>>>(q, off_w1, off_b1, off_w2, vs);

    // grid sample kv
    gsample_kernel<<<(BDIM * NDS * CDIM + 255) / 256, 256, 0, stream>>>(x2n, vs, kv);

    // k / v projections
    kvproj_kernel<<<BDIM * NDS, INNERC, 0, stream>>>(kv, k_w, v_w, kh, vT);

    // CPB bias (WMMA middle layer)
    cpb_wmma_kernel<<<BDIM * GG * NP, 256, 0, stream>>>(vs, cpb_w1, cpb_b1, cpb_w2, cpb_b2,
                                                        cpb_w3, cpb_b3, biasb);

    // sim = (q*s) k^T + bias   [per (b,h): M=1024, N=256, K=64]
    {
        dim3 grid(NP / 16, NDS / 64, BDIM * HH);
        wmma_gemm_kernel<DHH, true, false><<<grid, 32, 0, stream>>>(
            qh, kh, sim, biasb, nullptr,
            INNERC, INNERC, NDS, HH,
            (long long)NP * INNERC, 64LL,
            (long long)NDS * INNERC, 64LL,
            (long long)HH * NP * NDS, (long long)NP * NDS);
    }

    // softmax -> attn (f32 in d_out) + attn f16
    softmax_kernel<<<BDIM * HH * NP, NDS, 0, stream>>>(sim, attn, attnh);

    // out2 = attn * v   [per (b,h): M=1024, N=64, K=256] -> (B, N, INNER)
    {
        dim3 grid(NP / 16, DHH / 64, BDIM * HH);
        wmma_gemm_kernel<NDS, false, false><<<grid, 32, 0, stream>>>(
            attnh, vT, out2, nullptr, nullptr,
            NDS, NDS, INNERC, HH,
            (long long)HH * NP * NDS, (long long)NP * NDS,
            (long long)INNERC * NDS, (long long)DHH * NDS,
            (long long)NP * INNERC, 64LL);
    }

    // output projection: x = x1 + out2 @ out_w^T + out_b  [M=1024, N=128, K=512]
    f32_to_f16_kernel<<<(BDIM * NP * INNERC + 255) / 256, 256, 0, stream>>>(
        out2, out2h, BDIM * NP * INNERC);
    f32_to_f16_kernel<<<(CDIM * INNERC + 255) / 256, 256, 0, stream>>>(
        out_w, outwh, CDIM * INNERC);
    {
        dim3 grid(NP / 16, CDIM / 64, BDIM);
        wmma_gemm_kernel<INNERC, true, true><<<grid, 32, 0, stream>>>(
            out2h, outwh, xbuf, x1, out_b,
            INNERC, INNERC, CDIM, 1,
            (long long)NP * INNERC, 0LL, 0LL, 0LL,
            (long long)NP * CDIM, 0LL);
    }

    // Gram matrix x x^T for DPC-KNN  [per b: M=N=1024, K=128]
    f32_to_f16_kernel<<<(BDIM * NP * CDIM + 255) / 256, 256, 0, stream>>>(
        xbuf, xh, BDIM * NP * CDIM);
    {
        dim3 grid(NP / 16, NP / 64, BDIM);
        wmma_gemm_kernel<CDIM, false, false><<<grid, 32, 0, stream>>>(
            xh, xh, gram, nullptr, nullptr,
            CDIM, CDIM, NP, 1,
            (long long)NP * CDIM, 0LL,
            (long long)NP * CDIM, 0LL,
            (long long)NP * NP, 0LL);
    }

    // clustering
    rowstats_kernel<<<(BDIM * NP + 255) / 256, 256, 0, stream>>>(xbuf, score_w, score_b, xx, tw);
    dist_kernel<<<BDIM * NP, 256, 0, stream>>>(gram, xx, dist, density, dmax);
    distind_kernel<<<BDIM * NP, 256, 0, stream>>>(dist, density, dmax, score);
    topk_kernel<<<BDIM, 256, 0, stream>>>(score, idxdn);
    assign_kernel<<<(BDIM * NP + 255) / 256, 256, 0, stream>>>(dist, idxdn, cluster);
    center_override_kernel<<<1, 256, 0, stream>>>(idxdn, cluster);
    accw_kernel<<<(BDIM * NP + 255) / 256, 256, 0, stream>>>(cluster, tw, allw);
    merge_kernel<<<BDIM * NP, CDIM, 0, stream>>>(xbuf, cluster, tw, allw, xmerged);
}
